// GAT_14774687498698
// MI455X (gfx1250) — matrix operand order repaired and checked by reading, not compiled, not run on hardware
//
#include <hip/hip_runtime.h>
#include <hip/hip_bf16.h>

// ---------------------------------------------------------------------------
// GAT 2-layer forward for MI455X (gfx1250, wave32).
// GEMMs: v_wmma_f32_16x16x32_bf16, B (weights) register-resident across the
// whole K extent, A streamed with grid-stride row tiles + global prefetch.
// Edge phase: coalesced f32 atomics (HBM-bound part, ~optimal traffic).
// ---------------------------------------------------------------------------

typedef __attribute__((ext_vector_type(16))) __bf16 v16bf;
typedef __attribute__((ext_vector_type(8)))  float  v8f;

#define HEADS 4
#define HID   64
#define D1    256   // HEADS*HID
#define NCLS  10
#define NGRAPH 64

static __device__ __forceinline__ unsigned ordf(float f) {
  unsigned u = __float_as_uint(f);
  return (u & 0x80000000u) ? ~u : (u | 0x80000000u);
}
static __device__ __forceinline__ float unordf(unsigned u) {
  return (u & 0x80000000u) ? __uint_as_float(u & 0x7FFFFFFFu)
                           : __uint_as_float(~u);
}

// ---------------------------------------------------------------------------
// WMMA GEMM: H[N,256] = X[N,K] @ W[K,256], bf16 inputs / f32 accumulate.
// grid = (16 col tiles, ROWBLK row blocks), block = 32 (one wave).
// Per wave: preload ALL K/32 B fragments for its 16-col slice into registers
// (K=256 -> 64 VGPRs of packed bf16), then grid-stride over 16-row tiles,
// issuing K/32 back-to-back WMMAs per tile. A fragment per ISA 7.12.2:
// lane = row (lane%16); k = (e&7) + 8*half + 16*(e>=8) -> two contiguous
// 8-float runs per lane -> b128 loads. B: lane = col; same k as A.
// C/D: acc[r] -> row r + 8*half, col lane%16.
// ---------------------------------------------------------------------------
template <int K>
__global__ void gat_gemm_wmma(const float* __restrict__ X,
                              const float* __restrict__ W,
                              float* __restrict__ H,
                              int Ntiles) {
  constexpr int KC = K / 32;               // K chunks (4 or 8)
  const int lane = threadIdx.x;            // 0..31
  const int half = lane >> 4;
  const int lr   = lane & 15;
  const int col  = blockIdx.x * 16 + lr;   // B/C column for this lane

  // ---- Preload B fragments for the whole K extent (register resident) ----
  v16bf bfrag[KC];
#pragma unroll
  for (int c = 0; c < KC; ++c) {
#pragma unroll
    for (int e = 0; e < 16; ++e) {
      int kb = c * 32 + (e & 7) + half * 8 + (e >> 3) * 16;
      bfrag[c][e] = (__bf16)W[(size_t)kb * D1 + col];
    }
  }

  // ---- Stream row tiles ----
  for (int tr = blockIdx.y; tr < Ntiles; tr += gridDim.y) {
    const float* __restrict__ Xrow = X + (size_t)(tr * 16 + lr) * K;

    // Prefetch next tile's A stream for this lane (global_prefetch_b8).
    if (tr + (int)gridDim.y < Ntiles)
      __builtin_prefetch(X + (size_t)((tr + gridDim.y) * 16 + lr) * K, 0, 0);

    v8f acc = {};
#pragma unroll
    for (int c = 0; c < KC; ++c) {
      const float* base = Xrow + c * 32 + half * 8;
      float4 q0 = *(const float4*)(base + 0);
      float4 q1 = *(const float4*)(base + 4);
      float4 q2 = *(const float4*)(base + 16);
      float4 q3 = *(const float4*)(base + 20);
      float a16[16] = {q0.x, q0.y, q0.z, q0.w, q1.x, q1.y, q1.z, q1.w,
                       q2.x, q2.y, q2.z, q2.w, q3.x, q3.y, q3.z, q3.w};
      v16bf av;
#pragma unroll
      for (int e = 0; e < 16; ++e) av[e] = (__bf16)a16[e];

      acc = __builtin_amdgcn_wmma_f32_16x16x32_bf16(
          false, av, false, bfrag[c], (short)0, acc, false, false);
    }

    float* Hp = H + (size_t)tr * 16 * D1 + col;
#pragma unroll
    for (int r = 0; r < 8; ++r)
      Hp[(size_t)(r + half * 8) * D1] = acc[r];
  }
}

// ---------------------------------------------------------------------------
// Per-node attention coefficients: as/ad[n,h] = sum_c H[n,h,c] * a[h,c]
// ---------------------------------------------------------------------------
__global__ void gat_attn_coef(const float* __restrict__ H,
                              const float* __restrict__ a_src,
                              const float* __restrict__ a_dst,
                              float* __restrict__ as, float* __restrict__ ad,
                              int N) {
  int i = blockIdx.x * blockDim.x + threadIdx.x;   // i = n*HEADS + h
  if (i >= N * HEADS) return;
  int h = i & (HEADS - 1);
  const float* hp = H + (size_t)(i >> 2) * D1 + h * HID;
  float s = 0.f, d = 0.f;
#pragma unroll 8
  for (int c = 0; c < HID; ++c) {
    float v = hp[c];
    s += v * a_src[h * HID + c];
    d += v * a_dst[h * HID + c];
  }
  as[i] = s;
  ad[i] = d;
}

// ---------------------------------------------------------------------------
// Fills
// ---------------------------------------------------------------------------
__global__ void gat_fill_u32(unsigned* p, unsigned v, int n) {
  int i = blockIdx.x * blockDim.x + threadIdx.x;
  if (i < n) p[i] = v;
}
__global__ void gat_fill_f32(float* p, float v, int n) {
  int i = blockIdx.x * blockDim.x + threadIdx.x;
  if (i < n) p[i] = v;
}
__global__ void gat_bias_init(float* __restrict__ out,
                              const float* __restrict__ b, int n) {
  int i = blockIdx.x * blockDim.x + threadIdx.x;
  if (i < n) out[i] = b[i & (D1 - 1)];
}

// ---------------------------------------------------------------------------
// Edge pass A: e = leaky_relu(as[src]+ad[dst]); segment max via ordered-uint
// atomicMax. Self-loops appended for e >= E (src = dst = e - E), which also
// guarantees every dst segment is non-empty (m always written).
// ---------------------------------------------------------------------------
__global__ void gat_edge_logit(const int* __restrict__ ei,
                               const float* __restrict__ as,
                               const float* __restrict__ ad,
                               unsigned* __restrict__ m,
                               float* __restrict__ ebuf,
                               int E, int Etot) {
  int i = blockIdx.x * blockDim.x + threadIdx.x;
  if (i >= Etot * HEADS) return;
  int e = i >> 2, h = i & (HEADS - 1);
  int s, d;
  if (e < E) { s = ei[e]; d = ei[E + e]; } else { s = d = e - E; }
  float v = as[s * HEADS + h] + ad[d * HEADS + h];
  v = (v > 0.f) ? v : 0.2f * v;        // leaky_relu slope 0.2
  ebuf[i] = v;
  atomicMax(m + d * HEADS + h, ordf(v));
}

// Edge pass B: ex = exp(e - m[dst]); denom[dst] += ex (stable softmax)
__global__ void gat_edge_exp(const int* __restrict__ ei,
                             const unsigned* __restrict__ m,
                             float* __restrict__ ebuf,
                             float* __restrict__ denom,
                             int E, int Etot) {
  int i = blockIdx.x * blockDim.x + threadIdx.x;
  if (i >= Etot * HEADS) return;
  int e = i >> 2, h = i & (HEADS - 1);
  int d = (e < E) ? ei[E + e] : e - E;
  float ex = expf(ebuf[i] - unordf(m[d * HEADS + h]));
  ebuf[i] = ex;
  atomicAdd(denom + d * HEADS + h, ex);
}

// Edge pass C: out[dst,:] += alpha * H[src,:]. 64 threads per edge, each
// handles feature f0, f0+64, f0+128, f0+192 (head = j) -> coalesced gathers
// of H[src] and coalesced global_atomic_add_f32 into out[dst].
__global__ void gat_edge_scatter(const int* __restrict__ ei,
                                 const float* __restrict__ H,
                                 const float* __restrict__ ebuf,
                                 const float* __restrict__ denom,
                                 float* __restrict__ out,
                                 int E, int Etot) {
  int t = blockIdx.x * blockDim.x + threadIdx.x;
  int e = t >> 6;
  if (e >= Etot) return;
  int f0 = t & 63;
  int s, d;
  if (e < E) { s = ei[e]; d = ei[E + e]; } else { s = d = e - E; }
  const float* hs = H + (size_t)s * D1;
  float* od = out + (size_t)d * D1;
#pragma unroll
  for (int j = 0; j < HEADS; ++j) {
    float alpha = ebuf[e * HEADS + j] / (denom[d * HEADS + j] + 1e-16f);
    int f = j * HID + f0;
    atomicAdd(od + f, alpha * hs[f]);
  }
}

__global__ void gat_elu(float* p, int n) {
  int i = blockIdx.x * blockDim.x + threadIdx.x;
  if (i < n) {
    float v = p[i];
    p[i] = (v > 0.f) ? v : expm1f(v);
  }
}

// Global mean pool (segment sum + counts)
__global__ void gat_pool(const float* __restrict__ h,
                         const int* __restrict__ batch,
                         float* __restrict__ sums, float* __restrict__ cnt,
                         int N) {
  int t = blockIdx.x * blockDim.x + threadIdx.x;
  if (t >= N * D1) return;
  int n = t >> 8, f = t & (D1 - 1);
  int g = batch[n];
  atomicAdd(&sums[(size_t)g * D1 + f], h[t]);
  if (f == 0) atomicAdd(&cnt[g], 1.0f);
}

// Final linear + log_softmax: one thread per graph
__global__ void gat_head(const float* __restrict__ sums,
                         const float* __restrict__ cnt,
                         const float* __restrict__ lw,
                         const float* __restrict__ lb,
                         float* __restrict__ out) {
  int g = threadIdx.x;
  if (g >= NGRAPH) return;
  float inv = 1.0f / fmaxf(cnt[g], 1.0f);
  float logits[NCLS];
  float mx = -3.0e38f;
#pragma unroll
  for (int c = 0; c < NCLS; ++c) {
    float acc = lb[c];
    for (int k = 0; k < D1; ++k)
      acc += sums[(size_t)g * D1 + k] * inv * lw[k * NCLS + c];
    logits[c] = acc;
    mx = fmaxf(mx, acc);
  }
  float se = 0.f;
#pragma unroll
  for (int c = 0; c < NCLS; ++c) se += expf(logits[c] - mx);
  float lse = mx + logf(se);
#pragma unroll
  for (int c = 0; c < NCLS; ++c) out[g * NCLS + c] = logits[c] - lse;
}

// ---------------------------------------------------------------------------
// Launch
// ---------------------------------------------------------------------------
static inline int cdiv(long long a, long long b) { return (int)((a + b - 1) / b); }

extern "C" void kernel_launch(void* const* d_in, const int* in_sizes, int n_in,
                              void* d_out, int out_size, void* d_ws, size_t ws_size,
                              hipStream_t stream) {
  const int F_IN = 128;
  const int N = in_sizes[0] / F_IN;        // 50000 (divisible by 16)
  const int E = in_sizes[1] / 2;           // 800000
  const int Etot = E + N;                  // with self-loops
  const int Ntiles = N / 16;

  const float* x        = (const float*)d_in[0];
  const int*   ei       = (const int*)d_in[1];
  const int*   batch    = (const int*)d_in[2];
  const float* W1       = (const float*)d_in[3];
  const float* att_src1 = (const float*)d_in[4];
  const float* att_dst1 = (const float*)d_in[5];
  const float* b1       = (const float*)d_in[6];
  const float* W2       = (const float*)d_in[7];
  const float* att_src2 = (const float*)d_in[8];
  const float* att_dst2 = (const float*)d_in[9];
  const float* b2       = (const float*)d_in[10];
  const float* lin_w    = (const float*)d_in[11];
  const float* lin_b    = (const float*)d_in[12];
  float* out = (float*)d_out;

  // --- workspace carve (256B aligned) ---
  size_t off = 0;
  char* base = (char*)d_ws;
  auto carve = [&](size_t bytes) -> void* {
    void* p = base + off;
    off += (bytes + 255) & ~(size_t)255;
    return p;
  };
  float*    bufA  = (float*)carve((size_t)N * D1 * 4);   // H = X@W
  float*    bufB  = (float*)carve((size_t)N * D1 * 4);   // aggregation output
  float*    as    = (float*)carve((size_t)N * HEADS * 4);
  float*    ad    = (float*)carve((size_t)N * HEADS * 4);
  unsigned* mseg  = (unsigned*)carve((size_t)N * HEADS * 4);
  float*    denom = (float*)carve((size_t)N * HEADS * 4);
  float*    ebuf  = (float*)carve((size_t)Etot * HEADS * 4);
  float*    sums  = (float*)carve((size_t)NGRAPH * D1 * 4);
  float*    cnt   = (float*)carve((size_t)NGRAPH * 4);

  const int TB = 256;
  const int ROWBLK = 256;                  // row blocks per col tile (grid.y)

  // ====================== Layer 1 ======================
  gat_gemm_wmma<128><<<dim3(D1 / 16, ROWBLK), 32, 0, stream>>>(x, W1, bufA, Ntiles);
  gat_attn_coef<<<cdiv((long long)N * HEADS, TB), TB, 0, stream>>>(
      bufA, att_src1, att_dst1, as, ad, N);
  gat_fill_u32<<<cdiv((long long)N * HEADS, TB), TB, 0, stream>>>(mseg, 0u, N * HEADS);
  gat_fill_f32<<<cdiv((long long)N * HEADS, TB), TB, 0, stream>>>(denom, 0.f, N * HEADS);
  gat_bias_init<<<cdiv((long long)N * D1, TB), TB, 0, stream>>>(bufB, b1, N * D1);
  gat_edge_logit<<<cdiv((long long)Etot * HEADS, TB), TB, 0, stream>>>(
      ei, as, ad, mseg, ebuf, E, Etot);
  gat_edge_exp<<<cdiv((long long)Etot * HEADS, TB), TB, 0, stream>>>(
      ei, mseg, ebuf, denom, E, Etot);
  gat_edge_scatter<<<cdiv((long long)Etot * 64, TB), TB, 0, stream>>>(
      ei, bufA, ebuf, denom, bufB, E, Etot);
  gat_elu<<<cdiv((long long)N * D1, TB), TB, 0, stream>>>(bufB, N * D1);

  // ====================== Layer 2 ======================
  gat_gemm_wmma<256><<<dim3(D1 / 16, ROWBLK), 32, 0, stream>>>(bufB, W2, bufA, Ntiles);
  gat_attn_coef<<<cdiv((long long)N * HEADS, TB), TB, 0, stream>>>(
      bufA, att_src2, att_dst2, as, ad, N);
  gat_fill_u32<<<cdiv((long long)N * HEADS, TB), TB, 0, stream>>>(mseg, 0u, N * HEADS);
  gat_fill_f32<<<cdiv((long long)N * HEADS, TB), TB, 0, stream>>>(denom, 0.f, N * HEADS);
  gat_bias_init<<<cdiv((long long)N * D1, TB), TB, 0, stream>>>(bufB, b2, N * D1);
  gat_edge_logit<<<cdiv((long long)Etot * HEADS, TB), TB, 0, stream>>>(
      ei, as, ad, mseg, ebuf, E, Etot);
  gat_edge_exp<<<cdiv((long long)Etot * HEADS, TB), TB, 0, stream>>>(
      ei, mseg, ebuf, denom, E, Etot);
  gat_edge_scatter<<<cdiv((long long)Etot * 64, TB), TB, 0, stream>>>(
      ei, bufA, ebuf, denom, bufB, E, Etot);
  gat_elu<<<cdiv((long long)N * D1, TB), TB, 0, stream>>>(bufB, N * D1);

  // ====================== Pool + head ======================
  gat_fill_f32<<<cdiv((long long)NGRAPH * D1, TB), TB, 0, stream>>>(sums, 0.f, NGRAPH * D1);
  gat_fill_f32<<<1, NGRAPH, 0, stream>>>(cnt, 0.f, NGRAPH);
  gat_pool<<<cdiv((long long)N * D1, TB), TB, 0, stream>>>(bufB, batch, sums, cnt, N);
  gat_head<<<1, NGRAPH, 0, stream>>>(sums, cnt, lin_w, lin_b, out);
}
